// MHA_86457691668643
// MI455X (gfx1250) — compile-verified, matrix operand order checked by reading, not run on hardware
//
#include <hip/hip_runtime.h>

typedef __attribute__((ext_vector_type(16))) __bf16 bf16x16;
typedef __attribute__((ext_vector_type(8)))  float  f32x8;
typedef int v4i __attribute__((__vector_size__(16)));

#define N_EMBD 768
#define HEADS  12
#define HDIM   64
#define WIN    64
#define TSEQ   256
#define BATCH  256

#if defined(__has_builtin)
#if __has_builtin(__builtin_amdgcn_global_load_async_to_lds_b128) && \
    __has_builtin(__builtin_amdgcn_s_wait_asynccnt)
#define HAVE_ASYNC_LDS 1
#endif
#endif

__device__ __forceinline__ unsigned short f2bf(float f) {
  unsigned int u = __float_as_uint(f);
  unsigned int r = (u + 0x7fffu + ((u >> 16) & 1u)) >> 16;   // RNE
  return (unsigned short)r;
}
__device__ __forceinline__ unsigned int pkbf(float a, float b) {
  return (unsigned int)f2bf(a) | ((unsigned int)f2bf(b) << 16);
}
__device__ __forceinline__ f32x8 zero8() {
  f32x8 z; for (int i = 0; i < 8; i++) z[i] = 0.0f; return z;
}

// 16-byte global -> LDS copy: async DMA (ASYNCcnt) if available, else sync.
__device__ __forceinline__ void cp16_g2l(const unsigned short* g,
                                         unsigned short* l) {
#if HAVE_ASYNC_LDS
  __builtin_amdgcn_global_load_async_to_lds_b128(
      (__attribute__((address_space(1))) v4i*)g,
      (__attribute__((address_space(3))) v4i*)l, 0, 0);
#else
  *(uint4*)l = *(const uint4*)g;
#endif
}
__device__ __forceinline__ void async_wait_le4() {
#if HAVE_ASYNC_LDS
  __builtin_amdgcn_s_wait_asynccnt(4);
#endif
}
__device__ __forceinline__ void async_wait_0() {
#if HAVE_ASYNC_LDS
  __builtin_amdgcn_s_wait_asynccnt(0);
#endif
}

union FragU { bf16x16 v; uint4 q[2]; };

// 16-bit A/B fragment: two contiguous 16-byte chunks (K = half*8 and 16+half*8)
__device__ __forceinline__ bf16x16 frag_ld(const unsigned short* p0,
                                           const unsigned short* p1) {
  FragU u;
  u.q[0] = *(const uint4*)p0;
  u.q[1] = *(const uint4*)p1;
  return u.v;
}

// ---------------------------------------------------------------------------
// Kernel 0a: transpose-convert 4 weight matrices fp32 [K][N] -> bf16 [N][K]
// ---------------------------------------------------------------------------
__global__ __launch_bounds__(256)
void k_wprep(const float* __restrict__ wq, const float* __restrict__ wk,
             const float* __restrict__ wv, const float* __restrict__ wp,
             unsigned short* __restrict__ wt) {
  int idx = blockIdx.x * 256 + threadIdx.x;       // exact: 2304*256 = 768*768
  const float* src = (blockIdx.y == 0) ? wq : (blockIdx.y == 1) ? wk
                    : (blockIdx.y == 2) ? wv : wp;
  int k = idx / N_EMBD, n = idx % N_EMBD;
  wt[(size_t)blockIdx.y * (N_EMBD * N_EMBD) + (size_t)n * N_EMBD + k] =
      f2bf(src[idx]);
}

// ---------------------------------------------------------------------------
// Kernel 0b: convert activations x fp32 -> bf16 (one pass; enables raw async
// DMA staging in the GEMM and halves repeated x traffic)
// ---------------------------------------------------------------------------
__global__ __launch_bounds__(256)
void k_xprep(const float* __restrict__ x, unsigned short* __restrict__ xb) {
  size_t i = ((size_t)blockIdx.x * 256 + threadIdx.x) * 16;
  const float4* src = (const float4*)(x + i);
  float4 f0 = src[0], f1 = src[1], f2 = src[2], f3 = src[3];
  uint4 w0 = {pkbf(f0.x, f0.y), pkbf(f0.z, f0.w),
              pkbf(f1.x, f1.y), pkbf(f1.z, f1.w)};
  uint4 w1 = {pkbf(f2.x, f2.y), pkbf(f2.z, f2.w),
              pkbf(f3.x, f3.y), pkbf(f3.z, f3.w)};
  uint4* dst = (uint4*)(xb + i);
  dst[0] = w0; dst[1] = w1;
}

// ---------------------------------------------------------------------------
// Kernel A: fused QKV GEMM  [65536 x 768] @ [768 x 2304] -> Q,K bf16 [B,H,T,d]
// and V bf16 transposed [B,H,d,T].  Block tile 128x128, 8 waves of 32x64.
// Double-buffered async global->LDS staging overlaps DMA with WMMA.
// ---------------------------------------------------------------------------
__global__ __launch_bounds__(256)
void k_qkv(const unsigned short* __restrict__ xb,
           const unsigned short* __restrict__ wt,
           const float* __restrict__ bq, const float* __restrict__ bk,
           const float* __restrict__ bv,
           unsigned short* __restrict__ qo, unsigned short* __restrict__ ko,
           unsigned short* __restrict__ vto) {
  __shared__ __align__(16) unsigned short lds_a[2][128 * 40];
  __shared__ __align__(16) unsigned short lds_b[2][128 * 40];
  const int tid  = threadIdx.x;
  const int lane = tid & 31, wave = tid >> 5;
  const int half = lane >> 4, l16 = lane & 15;
  const int wm = wave & 3, wn = wave >> 2;          // 4 (M) x 2 (N) waves
  const int m0 = blockIdx.x * 128;
  const int n0 = blockIdx.y * 128;
  const int which = n0 / N_EMBD;                    // 0=Q 1=K 2=V (uniform)
  const int nn0 = n0 - which * N_EMBD;
  const unsigned short* wmat = wt + (size_t)which * (N_EMBD * N_EMBD);

  const int arow = tid >> 1;                        // 0..127
  const int akt  = (tid & 1) * 16;                  // element offset 0/16
  const unsigned short* asrc = xb   + (size_t)(m0 + arow) * N_EMBD + akt;
  const unsigned short* bsrc = wmat + (size_t)(nn0 + arow) * N_EMBD + akt;
  unsigned short* adst0 = &lds_a[0][arow * 40 + akt];
  unsigned short* bdst0 = &lds_b[0][arow * 40 + akt];
  unsigned short* adst1 = &lds_a[1][arow * 40 + akt];
  unsigned short* bdst1 = &lds_b[1][arow * 40 + akt];

  f32x8 acc[2][4];
  for (int i = 0; i < 2; i++)
    for (int j = 0; j < 4; j++) acc[i][j] = zero8();

  const int NIT = N_EMBD / 32;                      // 24
  // prologue: stage tile 0 into buffer 0
  cp16_g2l(asrc, adst0); cp16_g2l(asrc + 8, adst0 + 8);
  cp16_g2l(bsrc, bdst0); cp16_g2l(bsrc + 8, bdst0 + 8);

  for (int it = 0; it < NIT; ++it) {
    const int cur = it & 1;
    __syncthreads();    // everyone done computing from the buffer we refill next
    if (it + 1 < NIT) {
      const int kk = (it + 1) * 32;
      unsigned short* ad = cur ? adst0 : adst1;
      unsigned short* bd = cur ? bdst0 : bdst1;
      cp16_g2l(asrc + kk, ad); cp16_g2l(asrc + kk + 8, ad + 8);
      cp16_g2l(bsrc + kk, bd); cp16_g2l(bsrc + kk + 8, bd + 8);
      async_wait_le4();     // 4 ops just issued outstanding -> tile `it` landed
    } else {
      async_wait_0();
    }
    __syncthreads();

    bf16x16 afr[2], bfr[4];
    for (int mt = 0; mt < 2; mt++) {
      const unsigned short* base = &lds_a[cur][(wm * 32 + mt * 16 + l16) * 40];
      afr[mt] = frag_ld(base + half * 8, base + 16 + half * 8);
    }
    for (int nt = 0; nt < 4; nt++) {
      const unsigned short* base = &lds_b[cur][(wn * 64 + nt * 16 + l16) * 40];
      bfr[nt] = frag_ld(base + half * 8, base + 16 + half * 8);
    }
    for (int mt = 0; mt < 2; mt++)
      for (int nt = 0; nt < 4; nt++)
        acc[mt][nt] = __builtin_amdgcn_wmma_f32_16x16x32_bf16(
            false, afr[mt], false, bfr[nt], (short)0, acc[mt][nt], false, false);
  }

  const float* bias_p = (which == 0) ? bq : (which == 1) ? bk : bv;
  unsigned short* dst_qk = (which == 0) ? qo : ko;
  for (int nt = 0; nt < 4; nt++) {
    int nn = nn0 + wn * 64 + nt * 16 + l16;
    float bias = bias_p[nn];
    int h = nn >> 6, d = nn & 63;
    for (int mt = 0; mt < 2; mt++) {
      for (int r = 0; r < 8; r++) {
        int mg = m0 + wm * 32 + mt * 16 + r + 8 * half;  // D: m = r + 8*half
        int b = mg >> 8, t = mg & 255;
        unsigned short v16 = f2bf(acc[mt][nt][r] + bias);
        size_t bh = (size_t)b * HEADS + h;
        if (which == 2) vto[(bh * HDIM + d) * TSEQ + t] = v16;  // V^T
        else            dst_qk[(bh * TSEQ + t) * HDIM + d] = v16;
      }
    }
  }
}

// ---------------------------------------------------------------------------
// Kernel B: sliding-window attention per (b,h,chunk).  4 waves x 16 queries.
// ---------------------------------------------------------------------------
__global__ __launch_bounds__(128)
void k_attn(const unsigned short* __restrict__ q,
            const unsigned short* __restrict__ k,
            const unsigned short* __restrict__ vt,
            unsigned short* __restrict__ attn) {
  __shared__ __align__(16) unsigned short k_lds[128 * 72];   // [j][d]
  __shared__ __align__(16) unsigned short vt_lds[64 * 136];  // [d][j]
  __shared__ __align__(16) unsigned short p_lds[64 * 136];   // [i][j]
  const int tid  = threadIdx.x;
  const int lane = tid & 31, wave = tid >> 5;
  const int half = lane >> 4, l16 = lane & 15;
  int bi = blockIdx.x;
  int c = bi & 3, h = (bi >> 2) % HEADS, b = bi / (4 * HEADS);
  size_t bh = (size_t)b * HEADS + h;
  const unsigned short* qp = q  + bh * (TSEQ * HDIM);
  const unsigned short* kp = k  + bh * (TSEQ * HDIM);
  const unsigned short* vp = vt + bh * (HDIM * TSEQ);
  const int t0 = (c - 1) * WIN;                 // global t of concat key j=0

  { // stage K-cat rows (zeros for chunk 0 predecessor) - async DMA when valid
    int j = tid, t = t0 + j;
    unsigned short* dst = k_lds + j * 72;
    if (t < 0) {
      uint4 z = {0, 0, 0, 0};
      for (int i = 0; i < 8; i++) ((uint4*)dst)[i] = z;
    } else {
      const unsigned short* src = kp + (size_t)t * HDIM;
      for (int i = 0; i < 8; i++) cp16_g2l(src + i * 8, dst + i * 8);
    }
  }
  { // stage V^T rows: thread -> (d = tid/2, 64-col half)
    int d = tid >> 1, j0 = (tid & 1) * 64, t = t0 + j0;
    unsigned short* dst = vt_lds + d * 136 + j0;
    if (t < 0) {
      uint4 z = {0, 0, 0, 0};
      for (int i = 0; i < 8; i++) ((uint4*)dst)[i] = z;
    } else {
      const unsigned short* src = vp + (size_t)d * TSEQ + t;
      for (int i = 0; i < 8; i++) cp16_g2l(src + i * 8, dst + i * 8);
    }
  }
  async_wait_0();
  __syncthreads();

  const int qbase = wave * 16;
  // ---- S = Q @ K^T  (M=16, N=128, K=64) ----
  f32x8 s[8];
  for (int nt = 0; nt < 8; nt++) s[nt] = zero8();
  for (int ks = 0; ks < 2; ks++) {
    const unsigned short* qrow =
        qp + (size_t)(c * WIN + qbase + l16) * HDIM + ks * 32;
    bf16x16 afr = frag_ld(qrow + half * 8, qrow + 16 + half * 8);
    for (int nt = 0; nt < 8; nt++) {
      const unsigned short* base = k_lds + (nt * 16 + l16) * 72 + ks * 32;
      bf16x16 bfr = frag_ld(base + half * 8, base + 16 + half * 8);
      s[nt] = __builtin_amdgcn_wmma_f32_16x16x32_bf16(
          false, afr, false, bfr, (short)0, s[nt], false, false);
    }
  }
  // ---- mask + softmax (row m = r + 8*half, col j = nt*16 + l16) ----
  for (int r = 0; r < 8; r++) {
    int i = qbase + r + 8 * half;                    // query index in chunk
    float mx = -3.0e38f, vals[8];
    for (int nt = 0; nt < 8; nt++) {
      int j = nt * 16 + l16;
      bool valid = (j > i) && (j <= i + WIN) && ((c > 0) || (j >= WIN));
      float v = valid ? s[nt][r] * 0.125f : -1.0e30f; // 1/sqrt(64)
      vals[nt] = v;
      mx = fmaxf(mx, v);
    }
    mx = fmaxf(mx, __shfl_xor(mx, 1, 32));
    mx = fmaxf(mx, __shfl_xor(mx, 2, 32));
    mx = fmaxf(mx, __shfl_xor(mx, 4, 32));
    mx = fmaxf(mx, __shfl_xor(mx, 8, 32));
    float sum = 0.0f;
    for (int nt = 0; nt < 8; nt++) { vals[nt] = __expf(vals[nt] - mx); sum += vals[nt]; }
    sum += __shfl_xor(sum, 1, 32);
    sum += __shfl_xor(sum, 2, 32);
    sum += __shfl_xor(sum, 4, 32);
    sum += __shfl_xor(sum, 8, 32);
    float rinv = 1.0f / sum;
    for (int nt = 0; nt < 8; nt++)
      p_lds[(qbase + r + 8 * half) * 136 + nt * 16 + l16] =
          f2bf(vals[nt] * rinv);
  }
  __syncthreads();
  // ---- O = P @ Vcat  (M=16, N=64, K=128); B read from V^T (col-contig K) ----
  f32x8 o[4];
  for (int nt = 0; nt < 4; nt++) o[nt] = zero8();
  for (int ks = 0; ks < 4; ks++) {
    const unsigned short* prow = p_lds + (qbase + l16) * 136 + ks * 32;
    bf16x16 afr = frag_ld(prow + half * 8, prow + 16 + half * 8);
    for (int nt = 0; nt < 4; nt++) {
      const unsigned short* base = vt_lds + (nt * 16 + l16) * 136 + ks * 32;
      bf16x16 bfr = frag_ld(base + half * 8, base + 16 + half * 8);
      o[nt] = __builtin_amdgcn_wmma_f32_16x16x32_bf16(
          false, afr, false, bfr, (short)0, o[nt], false, false);
    }
  }
  for (int nt = 0; nt < 4; nt++) {
    int d = nt * 16 + l16;
    for (int r = 0; r < 8; r++) {
      int t = c * WIN + qbase + r + 8 * half;
      attn[((size_t)b * TSEQ + t) * N_EMBD + h * HDIM + d] = f2bf(o[nt][r]);
    }
  }
}

// ---------------------------------------------------------------------------
// Kernel C: output projection  attn(bf16) @ Wp^T(bf16,[N][K]) + bp -> fp32,
// same double-buffered async pipeline as kernel A.
// ---------------------------------------------------------------------------
__global__ __launch_bounds__(256)
void k_proj(const unsigned short* __restrict__ attn,
            const unsigned short* __restrict__ wtp,
            const float* __restrict__ bp, float* __restrict__ out) {
  __shared__ __align__(16) unsigned short lds_a[2][128 * 40];
  __shared__ __align__(16) unsigned short lds_b[2][128 * 40];
  const int tid  = threadIdx.x;
  const int lane = tid & 31, wave = tid >> 5;
  const int half = lane >> 4, l16 = lane & 15;
  const int wm = wave & 3, wn = wave >> 2;
  const int m0 = blockIdx.x * 128;
  const int n0 = blockIdx.y * 128;
  const int arow = tid >> 1;
  const int akt  = (tid & 1) * 16;
  const unsigned short* asrc = attn + (size_t)(m0 + arow) * N_EMBD + akt;
  const unsigned short* bsrc = wtp  + (size_t)(n0 + arow) * N_EMBD + akt;
  unsigned short* adst0 = &lds_a[0][arow * 40 + akt];
  unsigned short* bdst0 = &lds_b[0][arow * 40 + akt];
  unsigned short* adst1 = &lds_a[1][arow * 40 + akt];
  unsigned short* bdst1 = &lds_b[1][arow * 40 + akt];

  f32x8 acc[2][4];
  for (int i = 0; i < 2; i++)
    for (int j = 0; j < 4; j++) acc[i][j] = zero8();

  const int NIT = N_EMBD / 32;
  cp16_g2l(asrc, adst0); cp16_g2l(asrc + 8, adst0 + 8);
  cp16_g2l(bsrc, bdst0); cp16_g2l(bsrc + 8, bdst0 + 8);

  for (int it = 0; it < NIT; ++it) {
    const int cur = it & 1;
    __syncthreads();
    if (it + 1 < NIT) {
      const int kk = (it + 1) * 32;
      unsigned short* ad = cur ? adst0 : adst1;
      unsigned short* bd = cur ? bdst0 : bdst1;
      cp16_g2l(asrc + kk, ad); cp16_g2l(asrc + kk + 8, ad + 8);
      cp16_g2l(bsrc + kk, bd); cp16_g2l(bsrc + kk + 8, bd + 8);
      async_wait_le4();
    } else {
      async_wait_0();
    }
    __syncthreads();

    bf16x16 afr[2], bfr[4];
    for (int mt = 0; mt < 2; mt++) {
      const unsigned short* base = &lds_a[cur][(wm * 32 + mt * 16 + l16) * 40];
      afr[mt] = frag_ld(base + half * 8, base + 16 + half * 8);
    }
    for (int nt = 0; nt < 4; nt++) {
      const unsigned short* base = &lds_b[cur][(wn * 64 + nt * 16 + l16) * 40];
      bfr[nt] = frag_ld(base + half * 8, base + 16 + half * 8);
    }
    for (int mt = 0; mt < 2; mt++)
      for (int nt = 0; nt < 4; nt++)
        acc[mt][nt] = __builtin_amdgcn_wmma_f32_16x16x32_bf16(
            false, afr[mt], false, bfr[nt], (short)0, acc[mt][nt], false, false);
  }

  for (int nt = 0; nt < 4; nt++) {
    int n = n0 + wn * 64 + nt * 16 + l16;
    float bias = bp[n];
    for (int mt = 0; mt < 2; mt++)
      for (int r = 0; r < 8; r++) {
        int mg = m0 + wm * 32 + mt * 16 + r + 8 * half;
        out[(size_t)mg * N_EMBD + n] = acc[mt][nt][r] + bias;
      }
  }
}

// ---------------------------------------------------------------------------
extern "C" void kernel_launch(void* const* d_in, const int* in_sizes, int n_in,
                              void* d_out, int out_size, void* d_ws,
                              size_t ws_size, hipStream_t stream) {
  const float* x  = (const float*)d_in[0];
  const float* Wq = (const float*)d_in[1];
  const float* bq = (const float*)d_in[2];
  const float* Wk = (const float*)d_in[3];
  const float* bk = (const float*)d_in[4];
  const float* Wv = (const float*)d_in[5];
  const float* bv = (const float*)d_in[6];
  const float* Wp = (const float*)d_in[7];
  const float* bp = (const float*)d_in[8];
  float* out = (float*)d_out;

  const size_t WTN  = (size_t)N_EMBD * N_EMBD;              // 589824
  const size_t QKVN = (size_t)BATCH * HEADS * TSEQ * HDIM;  // 50331648
  unsigned short* ws16 = (unsigned short*)d_ws;
  unsigned short* wt   = ws16;              // 4 transposed bf16 weights
  unsigned short* xb   = ws16 + 4 * WTN;    // x as bf16 [B*T, C]
  unsigned short* qws  = xb + QKVN;         // Q  [B,H,T,d]
  unsigned short* kws  = qws + QKVN;        // K  [B,H,T,d]
  unsigned short* vtws = kws + QKVN;        // V^T[B,H,d,T]
  unsigned short* attn = vtws + QKVN;       // attn out [B,T,C] bf16

  k_wprep<<<dim3((N_EMBD * N_EMBD) / 256, 4), 256, 0, stream>>>(Wq, Wk, Wv, Wp, wt);
  k_xprep<<<dim3((int)(QKVN / (16 * 256))), 256, 0, stream>>>(x, xb);
  k_qkv<<<dim3((BATCH * TSEQ) / 128, (3 * N_EMBD) / 128), 256, 0, stream>>>(
      xb, wt, bq, bk, bv, qws, kws, vtws);
  k_attn<<<dim3(BATCH * HEADS * (TSEQ / WIN)), 128, 0, stream>>>(qws, kws, vtws, attn);
  k_proj<<<dim3((BATCH * TSEQ) / 128, N_EMBD / 128), 256, 0, stream>>>(
      attn, wt + 3 * WTN, bp, out);
}
